// DeepTDATransformer_74397423501976
// MI455X (gfx1250) — matrix operand order repaired and checked by reading, not compiled, not run on hardware
//
#include <hip/hip_runtime.h>
#include <hip/hip_bf16.h>
#include <math.h>

#define B_  4
#define S_  1000
#define IN_ 5
#define E_  256
#define L_  6
#define H_  8
#define NC_ 2
#define NL_ 50
#define DH_ 32
#define FF_ 1024
#define BS_ (B_*S_)

typedef __attribute__((ext_vector_type(16))) _Float16 v16h;
typedef __attribute__((ext_vector_type(8)))  _Float16 v8h;
typedef __attribute__((ext_vector_type(8)))  float    v8f;

// A-operand (16x32 f16): lane holds row M=lane&15; K indices are one
// contiguous 16-half run: half*16+{0..15}.
__device__ __forceinline__ v16h load_a_f16(const _Float16* __restrict__ p, int half) {
    v8h lo = *(const v8h*)(p + half * 16);
    v8h hi = *(const v8h*)(p + half * 16 + 8);
    return __builtin_shufflevector(lo, hi, 0, 1, 2, 3, 4, 5, 6, 7,
                                           8, 9, 10, 11, 12, 13, 14, 15);
}

// f32 -> f16 elementwise (weight/activation conversion)
__global__ void f2h_kernel(const float* __restrict__ src, _Float16* __restrict__ dst, int n) {
    int i = blockIdx.x * 256 + threadIdx.x;
    if (i < n) dst[i] = (_Float16)src[i];
}

// ---------------------------------------------------------------------------
// WMMA GEMM, all-f16 operands: C = act(A[M,K] @ W[N,K]^T + bias).
// One wave per 16x64 output tile; A loaded once per K-chunk, reused 4x.
// Writes f32 (Cf) and/or f16 (Ch) outputs.  act: 0=none, 1=relu, 2=gelu.
// ---------------------------------------------------------------------------
__global__ void wmma_gemm_kernel(const _Float16* __restrict__ A, int lda,
                                 const _Float16* __restrict__ Wm, int ldw,
                                 const float* __restrict__ bias,
                                 float* __restrict__ Cf, _Float16* __restrict__ Ch,
                                 int ldc, int K, int act) {
    int lane = threadIdx.x;
    int lcol = lane & 15, half = lane >> 4;
    int ra = blockIdx.x * 16 + lcol;   // A row this lane loads
    int nb = blockIdx.y * 64;          // first output column of this wave
    v8f acc[4] = {};
    for (int kc = 0; kc < K; kc += 32) {
        const _Float16* ap = A + (size_t)ra * lda + kc;
        if (kc + 32 < K) __builtin_prefetch(ap + 32, 0, 1);
        v16h av = load_a_f16(ap, half);
#pragma unroll
        for (int sub = 0; sub < 4; ++sub) {
            const _Float16* wp = Wm + (size_t)(nb + sub * 16 + lcol) * ldw + kc + half * 16;
            if (kc + 32 < K) __builtin_prefetch(wp + 32, 0, 1);
            v16h bv = *(const v16h*)wp;
            acc[sub] = __builtin_amdgcn_wmma_f32_16x16x32_f16(false, av, false, bv,
                                                              (short)0, acc[sub], false, false);
        }
    }
#pragma unroll
    for (int sub = 0; sub < 4; ++sub) {
#pragma unroll
        for (int v = 0; v < 8; ++v) {
            int r = blockIdx.x * 16 + v + 8 * half;
            int c = nb + sub * 16 + lcol;
            float val = acc[sub][v] + (bias ? bias[c] : 0.f);
            if (act == 1)      val = fmaxf(val, 0.f);
            else if (act == 2) val = 0.5f * val * (1.f + erff(val * 0.70710678118654752f));
            if (Cf) Cf[(size_t)r * ldc + c] = val;
            if (Ch) Ch[(size_t)r * ldc + c] = (_Float16)val;
        }
    }
}

// ---------------------------------------------------------------------------
// Flash-style attention on f16 Q/K/V: one wave per (b, h, 32-query tile).
// Two query sub-tiles share every K/V load: 8 WMMAs per 32-key chunk.
// scores = (Q K^T) * (ts[l,b]/sqrt(DH)); online softmax; out = P V (f16).
// ---------------------------------------------------------------------------
__global__ void attn_kernel(const _Float16* __restrict__ Q, const _Float16* __restrict__ Kp,
                            const _Float16* __restrict__ V, const float* __restrict__ tsv,
                            _Float16* __restrict__ O, int l) {
    __shared__ __align__(32) _Float16 Pl[2][16][32];
    int lane = threadIdx.x, lcol = lane & 15, half = lane >> 4;
    int qt = blockIdx.x, h = blockIdx.y, b = blockIdx.z;
    float scale = 0.17677669529663687f * tsv[l * B_ + b];  // 1/sqrt(32) * ts
    int q0 = qt * 32;

    v16h aq0, aq1;
    {
        int r0 = q0 + lcol;      if (r0 >= S_) r0 = S_ - 1;
        int r1 = q0 + 16 + lcol; if (r1 >= S_) r1 = S_ - 1;
        aq0 = load_a_f16(Q + ((size_t)(b * S_ + r0)) * E_ + h * DH_, half);
        aq1 = load_a_f16(Q + ((size_t)(b * S_ + r1)) * E_ + h * DH_, half);
    }
    float m0[8], l0[8], m1[8], l1[8];
    v8f acc00 = {}, acc01 = {}, acc10 = {}, acc11 = {};
#pragma unroll
    for (int v = 0; v < 8; ++v) { m0[v] = -3.0e38f; m1[v] = -3.0e38f; l0[v] = 0.f; l1[v] = 0.f; }

    for (int kt = 0; kt < S_; kt += 32) {
        v16h bk0, bk1;
        {
            int k0 = kt + lcol;      int kk0 = k0 < S_ ? k0 : S_ - 1;
            int k1 = kt + 16 + lcol; int kk1 = k1 < S_ ? k1 : S_ - 1;
            const _Float16* kp0 = Kp + ((size_t)(b * S_ + kk0)) * E_ + h * DH_ + half * 16;
            const _Float16* kp1 = Kp + ((size_t)(b * S_ + kk1)) * E_ + h * DH_ + half * 16;
            if (kt + 32 < S_) {
                __builtin_prefetch(kp0 + (size_t)32 * E_, 0, 1);
                __builtin_prefetch(kp1 + (size_t)32 * E_, 0, 1);
            }
            bk0 = *(const v16h*)kp0;
            bk1 = *(const v16h*)kp1;
        }
        v8f s00 = {}, s01 = {}, s10 = {}, s11 = {};
        s00 = __builtin_amdgcn_wmma_f32_16x16x32_f16(false, aq0, false, bk0, (short)0, s00, false, false);
        s01 = __builtin_amdgcn_wmma_f32_16x16x32_f16(false, aq0, false, bk1, (short)0, s01, false, false);
        s10 = __builtin_amdgcn_wmma_f32_16x16x32_f16(false, aq1, false, bk0, (short)0, s10, false, false);
        s11 = __builtin_amdgcn_wmma_f32_16x16x32_f16(false, aq1, false, bk1, (short)0, s11, false, false);
        bool mask0 = (kt + lcol) >= S_;
        bool mask1 = (kt + 16 + lcol) >= S_;
#pragma unroll
        for (int v = 0; v < 8; ++v) {
            // query sub-tile 0
            {
                float a0 = mask0 ? -1.0e30f : s00[v] * scale;
                float a1 = mask1 ? -1.0e30f : s01[v] * scale;
                float mx = fmaxf(a0, a1);
                for (int off = 8; off; off >>= 1)
                    mx = fmaxf(mx, __shfl_xor(mx, off, 32));
                float mn = fmaxf(m0[v], mx);
                float sc = __expf(m0[v] - mn);
                float p0 = __expf(a0 - mn);
                float p1 = __expf(a1 - mn);
                float rs = p0 + p1;
                for (int off = 8; off; off >>= 1)
                    rs += __shfl_xor(rs, off, 32);
                l0[v] = l0[v] * sc + rs;
                acc00[v] *= sc; acc01[v] *= sc;
                m0[v] = mn;
                Pl[0][v + 8 * half][lcol]      = (_Float16)p0;
                Pl[0][v + 8 * half][16 + lcol] = (_Float16)p1;
            }
            // query sub-tile 1
            {
                float a0 = mask0 ? -1.0e30f : s10[v] * scale;
                float a1 = mask1 ? -1.0e30f : s11[v] * scale;
                float mx = fmaxf(a0, a1);
                for (int off = 8; off; off >>= 1)
                    mx = fmaxf(mx, __shfl_xor(mx, off, 32));
                float mn = fmaxf(m1[v], mx);
                float sc = __expf(m1[v] - mn);
                float p0 = __expf(a0 - mn);
                float p1 = __expf(a1 - mn);
                float rs = p0 + p1;
                for (int off = 8; off; off >>= 1)
                    rs += __shfl_xor(rs, off, 32);
                l1[v] = l1[v] * sc + rs;
                acc10[v] *= sc; acc11[v] *= sc;
                m1[v] = mn;
                Pl[1][v + 8 * half][lcol]      = (_Float16)p0;
                Pl[1][v + 8 * half][16 + lcol] = (_Float16)p1;
            }
        }
        __syncthreads();
        v16h ap0 = load_a_f16(&Pl[0][lcol][0], half);
        v16h ap1 = load_a_f16(&Pl[1][lcol][0], half);
        v16h bv0, bv1;
#pragma unroll
        for (int e = 0; e < 16; ++e) {
            int key = kt + half * 16 + e; if (key >= S_) key = S_ - 1;
            const _Float16* vp = V + ((size_t)(b * S_ + key)) * E_ + h * DH_;
            bv0[e] = vp[lcol];
            bv1[e] = vp[16 + lcol];
        }
        acc00 = __builtin_amdgcn_wmma_f32_16x16x32_f16(false, ap0, false, bv0, (short)0, acc00, false, false);
        acc01 = __builtin_amdgcn_wmma_f32_16x16x32_f16(false, ap0, false, bv1, (short)0, acc01, false, false);
        acc10 = __builtin_amdgcn_wmma_f32_16x16x32_f16(false, ap1, false, bv0, (short)0, acc10, false, false);
        acc11 = __builtin_amdgcn_wmma_f32_16x16x32_f16(false, ap1, false, bv1, (short)0, acc11, false, false);
        __syncthreads();
    }
#pragma unroll
    for (int v = 0; v < 8; ++v) {
        int row0 = q0 + v + 8 * half;
        if (row0 < S_) {
            float inv = 1.f / l0[v];
            _Float16* op = O + ((size_t)(b * S_ + row0)) * E_ + h * DH_;
            op[lcol]      = (_Float16)(acc00[v] * inv);
            op[16 + lcol] = (_Float16)(acc01[v] * inv);
        }
        int row1 = q0 + 16 + v + 8 * half;
        if (row1 < S_) {
            float inv = 1.f / l1[v];
            _Float16* op = O + ((size_t)(b * S_ + row1)) * E_ + h * DH_;
            op[lcol]      = (_Float16)(acc10[v] * inv);
            op[16 + lcol] = (_Float16)(acc11[v] * inv);
        }
    }
}

// ---------------------------------------------------------------------------
// LayerNorm over E=256: out = LN(a [+ resid]) * g + beta [+ post[row%S]];
// writes f32 and optional f16 copy for downstream WMMA consumers.
// ---------------------------------------------------------------------------
__global__ void ln_kernel(const float* __restrict__ a, const float* __restrict__ resid,
                          const float* __restrict__ g, const float* __restrict__ beta,
                          const float* __restrict__ post, float* __restrict__ out,
                          _Float16* __restrict__ outh) {
    __shared__ float red[E_];
    int row = blockIdx.x, t = threadIdx.x;
    float v = a[(size_t)row * E_ + t];
    if (resid) v += resid[(size_t)row * E_ + t];
    red[t] = v; __syncthreads();
    for (int s = 128; s; s >>= 1) { if (t < s) red[t] += red[t + s]; __syncthreads(); }
    float mean = red[0] / E_; __syncthreads();
    float d = v - mean; red[t] = d * d; __syncthreads();
    for (int s = 128; s; s >>= 1) { if (t < s) red[t] += red[t + s]; __syncthreads(); }
    float var = red[0] / E_;
    float y = d * rsqrtf(var + 1e-5f) * g[t] + beta[t];
    if (post) y += post[(size_t)(row % S_) * E_ + t];
    out[(size_t)row * E_ + t] = y;
    if (outh) outh[(size_t)row * E_ + t] = (_Float16)y;
}

// Embedding first layer: h = relu(seq @ w1^T + b1), K=5 too small for WMMA.
__global__ void emb1_kernel(const float* __restrict__ seq, const float* __restrict__ w1,
                            const float* __restrict__ b1, _Float16* __restrict__ h) {
    int row = blockIdx.x, j = threadIdx.x;
    const float* sp = seq + (size_t)row * IN_;
    float acc = b1[j];
#pragma unroll
    for (int i = 0; i < IN_; ++i) acc += sp[i] * w1[j * IN_ + i];
    h[(size_t)row * 128 + j] = (_Float16)fmaxf(acc, 0.f);
}

// Point cloud means + landmark scores
__global__ void ph_points_kernel(const float* __restrict__ seq, const float* __restrict__ law,
                                 const float* __restrict__ lab,
                                 float* __restrict__ pm, float* __restrict__ score) {
    int i = blockIdx.x * blockDim.x + threadIdx.x;
    if (i >= BS_) return;
    int s = i % S_;
    float mv = 0.f;
#pragma unroll
    for (int kk = 0; kk < IN_; ++kk) mv += seq[(size_t)i * IN_ + kk];
    mv *= (1.f / IN_);
    pm[i] = mv;
    score[i] = law[0] * (float)s + law[1] * mv + lab[0];
}

// Iterative top-NL selection (descending value, tie -> lower index)
__global__ void topk_kernel(const float* __restrict__ score, const float* __restrict__ pm,
                            float* __restrict__ lmx, float* __restrict__ lmy) {
    __shared__ float sc[S_];
    __shared__ float bv[256]; __shared__ int bi[256];
    int b = blockIdx.x, t = threadIdx.x;
    for (int s = t; s < S_; s += 256) sc[s] = score[b * S_ + s];
    __syncthreads();
    for (int j = 0; j < NL_; ++j) {
        float best = -3.4e38f; int bidx = S_;
        for (int s = t; s < S_; s += 256) {
            float v = sc[s];
            if (v > best || (v == best && s < bidx)) { best = v; bidx = s; }
        }
        bv[t] = best; bi[t] = bidx; __syncthreads();
        for (int st = 128; st; st >>= 1) {
            if (t < st) {
                if (bv[t + st] > bv[t] || (bv[t + st] == bv[t] && bi[t + st] < bi[t])) {
                    bv[t] = bv[t + st]; bi[t] = bi[t + st];
                }
            }
            __syncthreads();
        }
        if (t == 0) {
            int w = bi[0];
            lmx[b * NL_ + j] = (float)w;
            lmy[b * NL_ + j] = pm[b * S_ + w];
            sc[w] = -3.4e38f;
        }
        __syncthreads();
    }
}

// Laplacian build + cyclic Jacobi eigensolver + 6 persistence features
__global__ void eig_kernel(const float* __restrict__ lmx, const float* __restrict__ lmy,
                           const float* __restrict__ fw, const float* __restrict__ db,
                           float* __restrict__ pf) {
    __shared__ float A[NL_ * NL_], Vm[NL_ * NL_];
    __shared__ float rsum[NL_];
    __shared__ float cs[8];
    int b = blockIdx.x, t = threadIdx.x;
    float fwa = fabsf(fw[0]), dbv = db[0];
    for (int idx = t; idx < NL_ * NL_; idx += 64) {
        int i = idx / NL_, j = idx % NL_;
        float dx = lmx[b * NL_ + i] - lmx[b * NL_ + j];
        float dy = lmy[b * NL_ + i] - lmy[b * NL_ + j];
        float d2 = dx * dx + dy * dy;
        float dist = d2 > 0.f ? sqrtf(d2) : 0.f;
        A[idx]  = __expf(-(dist * fwa + dbv));
        Vm[idx] = (i == j) ? 1.f : 0.f;
    }
    __syncthreads();
    if (t < NL_) {
        float s = 0.f;
        for (int j = 0; j < NL_; ++j) s += A[t * NL_ + j];
        rsum[t] = s;
    }
    __syncthreads();
    for (int idx = t; idx < NL_ * NL_; idx += 64) {
        int i = idx / NL_, j = idx % NL_;
        float kij = A[idx];
        A[idx] = (i == j) ? (rsum[i] - kij) : (-kij);
    }
    __syncthreads();
    for (int sweep = 0; sweep < 15; ++sweep) {
        for (int p = 0; p < NL_ - 1; ++p) {
            for (int q = p + 1; q < NL_; ++q) {
                if (t == 0) {
                    float apq = A[p * NL_ + q];
                    if (fabsf(apq) > 1e-10f) {
                        float app = A[p * NL_ + p], aqq = A[q * NL_ + q];
                        float tau = (aqq - app) / (2.f * apq);
                        float tt = (tau >= 0.f ? 1.f : -1.f) / (fabsf(tau) + sqrtf(1.f + tau * tau));
                        float c = rsqrtf(1.f + tt * tt);
                        float s = tt * c;
                        cs[0] = c; cs[1] = s; cs[2] = app; cs[3] = aqq; cs[4] = apq;
                    } else { cs[0] = 1.f; cs[1] = 0.f; }
                }
                __syncthreads();
                float c = cs[0], s = cs[1];
                if (s != 0.f && t < NL_) {
                    int i = t;
                    if (i == p) {
                        float app = cs[2], aqq = cs[3], apq = cs[4];
                        A[p * NL_ + p] = c * c * app - 2.f * s * c * apq + s * s * aqq;
                    } else if (i == q) {
                        float app = cs[2], aqq = cs[3], apq = cs[4];
                        A[q * NL_ + q] = s * s * app + 2.f * s * c * apq + c * c * aqq;
                        A[p * NL_ + q] = 0.f; A[q * NL_ + p] = 0.f;
                    } else {
                        float api = A[p * NL_ + i], aqi = A[q * NL_ + i];
                        float np_ = c * api - s * aqi;
                        float nq_ = s * api + c * aqi;
                        A[p * NL_ + i] = np_; A[i * NL_ + p] = np_;
                        A[q * NL_ + i] = nq_; A[i * NL_ + q] = nq_;
                    }
                    float vip = Vm[i * NL_ + p], viq = Vm[i * NL_ + q];
                    Vm[i * NL_ + p] = c * vip - s * viq;
                    Vm[i * NL_ + q] = s * vip + c * viq;
                }
                __syncthreads();
            }
        }
    }
    if (t == 0) {
        float d[NL_]; int ord[NL_];
        for (int i = 0; i < NL_; ++i) { d[i] = A[i * NL_ + i]; ord[i] = i; }
        for (int i = 1; i < NL_; ++i) {
            float dv = d[i]; int oi = ord[i]; int j = i - 1;
            while (j >= 0 && d[j] > dv) { d[j + 1] = d[j]; ord[j + 1] = ord[j]; --j; }
            d[j + 1] = dv; ord[j + 1] = oi;
        }
        float n0 = 0.f, s0 = 0.f;
        for (int i = 0; i < NL_; ++i) if (d[i] < 1e-6f) { n0 += 1.f; s0 += fabsf(d[i]); }
        float gap = d[1] - d[0];
        float mean4 = 0.25f * (d[2] + d[3] + d[4] + d[5]);
        float var4 = 0.f;
        for (int i = 2; i < 6; ++i) { float dd = d[i] - mean4; var4 += dd * dd; }
        float std4 = sqrtf(var4 / 3.f);
        int c2 = ord[1];
        float mu = 0.f;
        for (int i = 0; i < NL_; ++i) mu += Vm[i * NL_ + c2];
        mu /= NL_;
        float var = 0.f;
        for (int i = 0; i < NL_; ++i) { float dd = Vm[i * NL_ + c2] - mu; var += dd * dd; }
        float fied = sqrtf(var / (NL_ - 1));
        float* o = pf + b * 6;
        o[0] = n0; o[1] = s0; o[2] = gap; o[3] = fied; o[4] = mean4; o[5] = std4;
    }
}

// ts[l,b] = mean(sigmoid(relu(pf@w1^T+b1)@w2^T+b2))
__global__ void ts_kernel(const float* __restrict__ pf, const float* __restrict__ w1,
                          const float* __restrict__ b1, const float* __restrict__ w2,
                          const float* __restrict__ b2, float* __restrict__ ts) {
    __shared__ float h1[128]; __shared__ float red[256];
    int blk = blockIdx.x; int l = blk / B_; int b = blk % B_;
    int t = threadIdx.x;
    if (t < 128) {
        float acc = b1[l * 128 + t];
        for (int i = 0; i < 6; ++i) acc += pf[b * 6 + i] * w1[(size_t)l * 128 * 6 + t * 6 + i];
        h1[t] = fmaxf(acc, 0.f);
    }
    __syncthreads();
    float acc = b2[l * E_ + t];
    const float* wp = w2 + (size_t)l * E_ * 128 + (size_t)t * 128;
    for (int i = 0; i < 128; ++i) acc += h1[i] * wp[i];
    red[t] = 1.f / (1.f + __expf(-acc));
    __syncthreads();
    for (int s = 128; s; s >>= 1) { if (t < s) red[t] += red[t + s]; __syncthreads(); }
    if (t == 0) ts[l * B_ + b] = red[0] / E_;
}

// Softmax pooling over sequence
__global__ void pool_kernel(const float* __restrict__ x, float* __restrict__ pooled) {
    __shared__ float rs[S_];
    __shared__ float red[256];
    int b = blockIdx.x, t = threadIdx.x;
    for (int s = t; s < S_; s += 256) {
        const float* xp = x + ((size_t)(b * S_ + s)) * E_;
        float a = 0.f;
        for (int e = 0; e < E_; ++e) a += xp[e];
        rs[s] = a;
    }
    __syncthreads();
    float mx = -3.4e38f;
    for (int s = t; s < S_; s += 256) mx = fmaxf(mx, rs[s]);
    red[t] = mx; __syncthreads();
    for (int st = 128; st; st >>= 1) { if (t < st) red[t] = fmaxf(red[t], red[t + st]); __syncthreads(); }
    mx = red[0]; __syncthreads();
    float se = 0.f;
    for (int s = t; s < S_; s += 256) se += __expf(rs[s] - mx);
    red[t] = se; __syncthreads();
    for (int st = 128; st; st >>= 1) { if (t < st) red[t] += red[t + st]; __syncthreads(); }
    float inv = 1.f / red[0];
    __syncthreads();
    for (int s = t; s < S_; s += 256) rs[s] = __expf(rs[s] - mx) * inv;
    __syncthreads();
    float acc = 0.f;
    for (int s = 0; s < S_; ++s) acc += x[((size_t)(b * S_ + s)) * E_ + t] * rs[s];
    pooled[b * E_ + t] = acc;
}

// Final LN + MLP classifier
__global__ void cls_kernel(const float* __restrict__ pooled, const float* __restrict__ g,
                           const float* __restrict__ beta, const float* __restrict__ w1,
                           const float* __restrict__ b1, const float* __restrict__ w2,
                           const float* __restrict__ b2, float* __restrict__ out) {
    __shared__ float pl[E_]; __shared__ float z[128]; __shared__ float red[128];
    int b = blockIdx.x, t = threadIdx.x;  // 128 threads
    pl[t] = pooled[b * E_ + t]; pl[t + 128] = pooled[b * E_ + t + 128];
    __syncthreads();
    red[t] = pl[t] + pl[t + 128]; __syncthreads();
    for (int s = 64; s; s >>= 1) { if (t < s) red[t] += red[t + s]; __syncthreads(); }
    float mean = red[0] / E_; __syncthreads();
    float d0 = pl[t] - mean, d1 = pl[t + 128] - mean;
    red[t] = d0 * d0 + d1 * d1; __syncthreads();
    for (int s = 64; s; s >>= 1) { if (t < s) red[t] += red[t + s]; __syncthreads(); }
    float rstd = rsqrtf(red[0] / E_ + 1e-5f);
    __syncthreads();
    pl[t]       = d0 * rstd * g[t] + beta[t];
    pl[t + 128] = d1 * rstd * g[t + 128] + beta[t + 128];
    __syncthreads();
    float acc = b1[t];
    for (int i = 0; i < E_; ++i) acc += pl[i] * w1[t * E_ + i];
    z[t] = fmaxf(acc, 0.f);
    __syncthreads();
    if (t < NC_) {
        float a = b2[t];
        for (int i = 0; i < 128; ++i) a += z[i] * w2[t * 128 + i];
        out[b * NC_ + t] = a;
    }
}

extern "C" void kernel_launch(void* const* d_in, const int* in_sizes, int n_in,
                              void* d_out, int out_size, void* d_ws, size_t ws_size,
                              hipStream_t stream) {
    (void)in_sizes; (void)n_in; (void)out_size; (void)ws_size;
    const float* seq    = (const float*)d_in[0];
    const float* ph_fw  = (const float*)d_in[1];
    const float* ph_db  = (const float*)d_in[2];
    const float* ph_law = (const float*)d_in[3];
    const float* ph_lab = (const float*)d_in[4];
    const float* emb_w1 = (const float*)d_in[5];
    const float* emb_b1 = (const float*)d_in[6];
    const float* emb_w2 = (const float*)d_in[7];
    const float* emb_b2 = (const float*)d_in[8];
    const float* emb_lg = (const float*)d_in[9];
    const float* emb_lb = (const float*)d_in[10];
    const float* pos    = (const float*)d_in[11];
    const float* qw     = (const float*)d_in[12];
    const float* qb     = (const float*)d_in[13];
    const float* kw     = (const float*)d_in[14];
    const float* kb     = (const float*)d_in[15];
    const float* vw     = (const float*)d_in[16];
    const float* vb     = (const float*)d_in[17];
    const float* pe_w1  = (const float*)d_in[18];
    const float* pe_b1  = (const float*)d_in[19];
    const float* pe_w2  = (const float*)d_in[20];
    const float* pe_b2  = (const float*)d_in[21];
    const float* ow     = (const float*)d_in[22];
    const float* obv    = (const float*)d_in[23];
    const float* ln_g   = (const float*)d_in[24];
    const float* ln_b   = (const float*)d_in[25];
    const float* f_w1   = (const float*)d_in[26];
    const float* f_b1   = (const float*)d_in[27];
    const float* f_w2   = (const float*)d_in[28];
    const float* f_b2   = (const float*)d_in[29];
    const float* c_lg   = (const float*)d_in[30];
    const float* c_lb   = (const float*)d_in[31];
    const float* c_w1   = (const float*)d_in[32];
    const float* c_b1   = (const float*)d_in[33];
    const float* c_w2   = (const float*)d_in[34];
    const float* c_b2   = (const float*)d_in[35];
    float* out = (float*)d_out;

    // ---- f32 workspace ----
    float* Wf = (float*)d_ws;
    float* x      = Wf; Wf += (size_t)BS_ * E_;
    float* tmp    = Wf; Wf += (size_t)BS_ * E_;
    float* pm     = Wf; Wf += BS_;
    float* score  = Wf; Wf += BS_;
    float* lmx    = Wf; Wf += B_ * NL_;
    float* lmy    = Wf; Wf += B_ * NL_;
    float* pf     = Wf; Wf += B_ * 6;
    float* tsb    = Wf; Wf += L_ * B_;
    float* pooled = Wf; Wf += B_ * E_;
    // ---- f16 workspace (all region sizes are multiples of 16 halves) ----
    _Float16* Wh = (_Float16*)Wf;
    _Float16* xh   = Wh; Wh += (size_t)BS_ * E_;
    _Float16* qh   = Wh; Wh += (size_t)BS_ * E_;
    _Float16* kh   = Wh; Wh += (size_t)BS_ * E_;
    _Float16* vh   = Wh; Wh += (size_t)BS_ * E_;
    _Float16* atth = Wh; Wh += (size_t)BS_ * E_;
    _Float16* ffhh = Wh; Wh += (size_t)BS_ * FF_;
    _Float16* hh   = Wh; Wh += (size_t)BS_ * 128;
    _Float16* emb_w2h = Wh; Wh += (size_t)E_ * 128;
    _Float16* qwh  = Wh; Wh += (size_t)L_ * E_ * E_;
    _Float16* kwh  = Wh; Wh += (size_t)L_ * E_ * E_;
    _Float16* vwh  = Wh; Wh += (size_t)L_ * E_ * E_;
    _Float16* owh  = Wh; Wh += (size_t)L_ * E_ * E_;
    _Float16* fw1h = Wh; Wh += (size_t)L_ * FF_ * E_;
    _Float16* fw2h = Wh; Wh += (size_t)L_ * E_ * FF_;

    // ---- weight conversion (once per launch; deterministic) ----
    {
        int n;
        n = E_ * 128;          f2h_kernel<<<(n + 255) / 256, 256, 0, stream>>>(emb_w2, emb_w2h, n);
        n = L_ * E_ * E_;      f2h_kernel<<<(n + 255) / 256, 256, 0, stream>>>(qw, qwh, n);
        n = L_ * E_ * E_;      f2h_kernel<<<(n + 255) / 256, 256, 0, stream>>>(kw, kwh, n);
        n = L_ * E_ * E_;      f2h_kernel<<<(n + 255) / 256, 256, 0, stream>>>(vw, vwh, n);
        n = L_ * E_ * E_;      f2h_kernel<<<(n + 255) / 256, 256, 0, stream>>>(ow, owh, n);
        n = L_ * FF_ * E_;     f2h_kernel<<<(n + 255) / 256, 256, 0, stream>>>(f_w1, fw1h, n);
        n = L_ * E_ * FF_;     f2h_kernel<<<(n + 255) / 256, 256, 0, stream>>>(f_w2, fw2h, n);
    }

    // --- persistence branch ---
    ph_points_kernel<<<(BS_ + 255) / 256, 256, 0, stream>>>(seq, ph_law, ph_lab, pm, score);
    topk_kernel<<<B_, 256, 0, stream>>>(score, pm, lmx, lmy);
    eig_kernel<<<B_, 64, 0, stream>>>(lmx, lmy, ph_fw, ph_db, pf);
    ts_kernel<<<L_ * B_, 256, 0, stream>>>(pf, pe_w1, pe_b1, pe_w2, pe_b2, tsb);

    // --- embedding ---
    emb1_kernel<<<BS_, 128, 0, stream>>>(seq, emb_w1, emb_b1, hh);
    wmma_gemm_kernel<<<dim3(BS_ / 16, E_ / 64), 32, 0, stream>>>(
        hh, 128, emb_w2h, 128, emb_b2, tmp, nullptr, E_, 128, 0);
    ln_kernel<<<BS_, E_, 0, stream>>>(tmp, nullptr, emb_lg, emb_lb, pos, x, xh);

    // --- transformer layers ---
    for (int l = 0; l < L_; ++l) {
        wmma_gemm_kernel<<<dim3(BS_ / 16, E_ / 64), 32, 0, stream>>>(
            xh, E_, qwh + (size_t)l * E_ * E_, E_, qb + l * E_, nullptr, qh, E_, E_, 0);
        wmma_gemm_kernel<<<dim3(BS_ / 16, E_ / 64), 32, 0, stream>>>(
            xh, E_, kwh + (size_t)l * E_ * E_, E_, kb + l * E_, nullptr, kh, E_, E_, 0);
        wmma_gemm_kernel<<<dim3(BS_ / 16, E_ / 64), 32, 0, stream>>>(
            xh, E_, vwh + (size_t)l * E_ * E_, E_, vb + l * E_, nullptr, vh, E_, E_, 0);
        attn_kernel<<<dim3((S_ + 31) / 32, H_, B_), 32, 0, stream>>>(qh, kh, vh, tsb, atth, l);
        wmma_gemm_kernel<<<dim3(BS_ / 16, E_ / 64), 32, 0, stream>>>(
            atth, E_, owh + (size_t)l * E_ * E_, E_, obv + l * E_, tmp, nullptr, E_, E_, 0);
        ln_kernel<<<BS_, E_, 0, stream>>>(x, tmp, ln_g + l * E_, ln_b + l * E_, nullptr, x, xh);
        wmma_gemm_kernel<<<dim3(BS_ / 16, FF_ / 64), 32, 0, stream>>>(
            xh, E_, fw1h + (size_t)l * FF_ * E_, E_, f_b1 + l * FF_, nullptr, ffhh, FF_, E_, 2);
        wmma_gemm_kernel<<<dim3(BS_ / 16, E_ / 64), 32, 0, stream>>>(
            ffhh, FF_, fw2h + (size_t)l * E_ * FF_, FF_, f_b2 + l * E_, tmp, nullptr, E_, FF_, 0);
        ln_kernel<<<BS_, E_, 0, stream>>>(x, tmp, ln_g + l * E_, ln_b + l * E_, nullptr, x, xh);
    }

    // --- pooling + classifier ---
    pool_kernel<<<B_, 256, 0, stream>>>(x, pooled);
    cls_kernel<<<B_, 128, 0, stream>>>(pooled, c_lg, c_lb, c_w1, c_b1, c_w2, c_b2, out);
}
